// DilatedMSA_14370960572446
// MI455X (gfx1250) — compile-verified, matrix operand order checked by reading, not run on hardware
//
#include <hip/hip_runtime.h>
#include <hip/hip_bf16.h>

// ---------------------------------------------------------------------------
// Fused DilatedMSA for MI455X (gfx1250, wave32, WMMA bf16 16x16x32)
// One workgroup per (b,l) tile: QKV projection + 2-head attention fully in LDS.
// ---------------------------------------------------------------------------

typedef __attribute__((ext_vector_type(16))) __bf16 v16bf;
typedef __attribute__((ext_vector_type(2)))  __bf16 v2bf;
typedef __attribute__((ext_vector_type(8)))  float  v8f;
typedef __attribute__((ext_vector_type(4)))  float  v4f;
typedef __attribute__((ext_vector_type(2)))  float  v2f;
typedef __attribute__((ext_vector_type(4)))  int    v4i;
typedef __attribute__((ext_vector_type(8)))  int    v8i;

#define G_   256   // tokens per (b,l) tile
#define C_   128   // channels
#define H_   2     // heads
#define HD_  64    // head dim
#define RSCALE 0.0883883476483184f  // 1/sqrt(128)

__device__ __forceinline__ unsigned short f2bf(float f) {
  unsigned u = __builtin_bit_cast(unsigned, f);
  u += 0x7FFFu + ((u >> 16) & 1u);          // round-to-nearest-even
  return (unsigned short)(u >> 16);
}

// Packed f32x2 -> bf16x2 (v_cvt_pk_bf16_f32 when available).
__device__ __forceinline__ unsigned pk_bf16(float a, float b) {
#if __has_builtin(__builtin_amdgcn_cvt_pk_bf16_f32)
  v2bf r = __builtin_amdgcn_cvt_pk_bf16_f32(a, b);
  return __builtin_bit_cast(unsigned, r);
#else
  return (unsigned)f2bf(a) | ((unsigned)f2bf(b) << 16);
#endif
}

// Load a 16-bit WMMA fragment from LDS (row-major bf16 buffer).
// p points at &buf[row*stride + kbase + laneHi*8]; lane covers K offsets
// {laneHi*8 + 0..7} U {16 + laneHi*8 + 0..7} per the CDNA5 16-bit A/B layout.
__device__ __forceinline__ v16bf lds_frag16(const unsigned short* p) {
  v8i w;
  w.lo = *(const v4i*)(p);
  w.hi = *(const v4i*)(p + 16);
  return __builtin_bit_cast(v16bf, w);
}

// Load an A-fragment directly from global f32 (convert to bf16 on the fly).
// p points at &X[row][kbase + laneHi*8]; 32B-aligned.
__device__ __forceinline__ v16bf gmem_frag_f32(const float* p) {
  v4f a0 = *(const v4f*)(p);
  v4f a1 = *(const v4f*)(p + 4);
  v4f b0 = *(const v4f*)(p + 16);
  v4f b1 = *(const v4f*)(p + 20);
  v8i w;
  w[0] = (int)pk_bf16(a0[0], a0[1]); w[1] = (int)pk_bf16(a0[2], a0[3]);
  w[2] = (int)pk_bf16(a1[0], a1[1]); w[3] = (int)pk_bf16(a1[2], a1[3]);
  w[4] = (int)pk_bf16(b0[0], b0[1]); w[5] = (int)pk_bf16(b0[2], b0[3]);
  w[6] = (int)pk_bf16(b1[0], b1[1]); w[7] = (int)pk_bf16(b1[2], b1[3]);
  return __builtin_bit_cast(v16bf, w);
}

__device__ __forceinline__ v8f wmma_bf16(v16bf a, v16bf b, v8f c) {
  return __builtin_amdgcn_wmma_f32_16x16x32_bf16(
      /*neg_a=*/false, a, /*neg_b=*/false, b,
      /*c_mod=*/(short)0, c, /*reuse_a=*/false, /*reuse_b=*/false);
}

__global__ __launch_bounds__(256)
void dilated_msa_fused(const float* __restrict__ x,
                       const float* __restrict__ Wq,
                       const float* __restrict__ bq,
                       float* __restrict__ out) {
  const int tid  = threadIdx.x;
  const int wave = tid >> 5;
  const int lane = tid & 31;
  const int l16  = lane & 15;
  const int lhi  = lane >> 4;          // 0: lanes 0-15, 1: lanes 16-31
  const int bl   = blockIdx.x;

  const float* xt = x   + (size_t)bl * G_ * C_;
  float*       ot = out + (size_t)bl * G_ * C_;

  // LDS layout (bf16 shorts): 160 KB total.
  //   Qs  [256][64]  @ 0       (32 KB)   scaled Q, row-major
  //   Ks  [256][64]  @ 16384   (32 KB)   K, row-major
  //   Vt  [64][256]  @ 32768   (32 KB)   V transposed (dim-major)
  //   union @ 49152  (64 KB):  W chunk [192][128] (phase A, 48 KB)
  //                            P scratch, 8 waves x [16][256] (phase B, 64 KB)
  __shared__ unsigned short smem[81920];
  unsigned short* Qs = smem;
  unsigned short* Ks = smem + 16384;
  unsigned short* Vt = smem + 32768;
  unsigned short* Ws = smem + 49152;
  unsigned short* Ps = smem + 49152 + wave * (16 * 256);

  for (int h = 0; h < H_; ++h) {
    __syncthreads();  // previous head fully consumed before overwriting LDS

    // ---- Load W chunk for this head (rows: 64 Q, 64 K, 64 V) as bf16 ------
    // Column pairs: packed cvt + b32 LDS store.
    for (int idx = tid; idx < 192 * 64; idx += 256) {
      const int j   = idx >> 6;                       // local row 0..191
      const int cp  = (idx & 63) * 2;                 // column pair base
      const int seg = j >> 6;                         // 0=q, 1=k, 2=v
      const int grow = seg * C_ + h * HD_ + (j & 63); // row in W_qkv[384][128]
      const v2f wv = *(const v2f*)(Wq + (size_t)grow * C_ + cp);
      *(unsigned*)(Ws + j * C_ + cp) = pk_bf16(wv[0], wv[1]);
    }
    __syncthreads();

    // ---- Phase A: QKV = X @ W^T + b (M=256, N=192, K=128) ------------------
    {
      const int mt0 = wave * 2;  // this wave owns M-tiles {mt0, mt0+1}
      v16bf afr[2][4];
#pragma unroll
      for (int mi = 0; mi < 2; ++mi) {
        const int tok = (mt0 + mi) * 16 + l16;
        const float* rowp = xt + (size_t)tok * C_;
        __builtin_prefetch(rowp, 0, 0);               // global_prefetch_b8
#pragma unroll
        for (int kf = 0; kf < 4; ++kf)
          afr[mi][kf] = gmem_frag_f32(rowp + kf * 32 + lhi * 8);
      }

      for (int nt = 0; nt < 12; ++nt) {
        const int seg  = nt >> 2;            // 0=q, 1=k, 2=v
        const int dimb = (nt & 3) * 16;      // head-dim base of this N-tile
        const unsigned short* wp = Ws + (seg * 64 + dimb + l16) * C_;
        v16bf bfr[4];
#pragma unroll
        for (int kf = 0; kf < 4; ++kf)
          bfr[kf] = lds_frag16(wp + kf * 32 + lhi * 8);
        const float bv = bq[seg * C_ + h * HD_ + dimb + l16];

#pragma unroll
        for (int mi = 0; mi < 2; ++mi) {
          v8f acc = {};
#pragma unroll
          for (int kf = 0; kf < 4; ++kf)
            acc = wmma_bf16(afr[mi][kf], bfr[kf], acc);

          const int tokb = (mt0 + mi) * 16 + lhi * 8; // D layout: M = r + 8*lhi
          if (seg == 0) {        // Q: fold 1/sqrt(C) into Q
#pragma unroll
            for (int r = 0; r < 8; ++r)
              Qs[(tokb + r) * HD_ + dimb + l16] = f2bf((acc[r] + bv) * RSCALE);
          } else if (seg == 1) { // K row-major
#pragma unroll
            for (int r = 0; r < 8; ++r)
              Ks[(tokb + r) * HD_ + dimb + l16] = f2bf(acc[r] + bv);
          } else {               // V transposed: Vt[dim][token]; consecutive r
                                 // are consecutive tokens -> packed b32 stores
#pragma unroll
            for (int r = 0; r < 8; r += 2)
              *(unsigned*)(Vt + (dimb + l16) * G_ + tokb + r) =
                  pk_bf16(acc[r] + bv, acc[r + 1] + bv);
          }
        }
      }
    }
    __syncthreads();

    // ---- Phase B: attention (per wave: M-tiles {wave, wave+8}) -------------
#pragma unroll
    for (int mi = 0; mi < 2; ++mi) {
      const int mt = wave + mi * 8;

      // Q A-fragments (head dim = 64 -> 2 K-fragments)
      const unsigned short* qp = Qs + (mt * 16 + l16) * HD_;
      v16bf qa0 = lds_frag16(qp + lhi * 8);
      v16bf qa1 = lds_frag16(qp + 32 + lhi * 8);

      // S = Q @ K^T : full 256-wide row strip in registers (16 x v8f)
      v8f s[16];
#pragma unroll
      for (int nt = 0; nt < 16; ++nt) s[nt] = (v8f){};
#pragma unroll
      for (int nt = 0; nt < 16; ++nt) {
        const unsigned short* kp = Ks + (nt * 16 + l16) * HD_;
        s[nt] = wmma_bf16(qa0, lds_frag16(kp + lhi * 8), s[nt]);
        s[nt] = wmma_bf16(qa1, lds_frag16(kp + 32 + lhi * 8), s[nt]);
      }

      // Softmax over N=256. Row M = r + 8*lhi lives in VGPR r across 16 lanes
      // of this half-wave, so reduce over regs then xor-shuffle masks 8,4,2,1.
#pragma unroll
      for (int r = 0; r < 8; ++r) {
        float m = s[0][r];
#pragma unroll
        for (int nt = 1; nt < 16; ++nt) m = fmaxf(m, s[nt][r]);
#pragma unroll
        for (int msk = 8; msk >= 1; msk >>= 1)
          m = fmaxf(m, __shfl_xor(m, msk, 32));
        float sum = 0.f;
#pragma unroll
        for (int nt = 0; nt < 16; ++nt) {
          const float e = __expf(s[nt][r] - m);
          s[nt][r] = e;
          sum += e;
        }
#pragma unroll
        for (int msk = 8; msk >= 1; msk >>= 1)
          sum += __shfl_xor(sum, msk, 32);
        const float inv = 1.0f / sum;
#pragma unroll
        for (int nt = 0; nt < 16; ++nt) s[nt][r] *= inv;
      }

      // P -> per-wave LDS scratch (D layout -> row-major [16][256] bf16)
#pragma unroll
      for (int nt = 0; nt < 16; ++nt)
#pragma unroll
        for (int r = 0; r < 8; ++r)
          Ps[(lhi * 8 + r) * G_ + nt * 16 + l16] = f2bf(s[nt][r]);
      asm volatile("s_wait_dscnt 0" ::: "memory");  // same-wave LDS RAW

      // O = P @ V  (M=16, N=64, K=256 -> 8 K-fragments x 4 N-tiles)
      v8f o[4];
#pragma unroll
      for (int n2 = 0; n2 < 4; ++n2) o[n2] = (v8f){};
#pragma unroll
      for (int kf = 0; kf < 8; ++kf) {
        v16bf pa = lds_frag16(Ps + l16 * G_ + kf * 32 + lhi * 8);
#pragma unroll
        for (int n2 = 0; n2 < 4; ++n2) {
          v16bf vb = lds_frag16(Vt + (n2 * 16 + l16) * G_ + kf * 32 + lhi * 8);
          o[n2] = wmma_bf16(pa, vb, o[n2]);
        }
      }

      // Store O (f32) to out[..., h*64 + dim]
#pragma unroll
      for (int n2 = 0; n2 < 4; ++n2)
#pragma unroll
        for (int r = 0; r < 8; ++r)
          ot[(size_t)(mt * 16 + lhi * 8 + r) * C_ + h * HD_ + n2 * 16 + l16] =
              o[n2][r];
    }
  }
}

extern "C" void kernel_launch(void* const* d_in, const int* in_sizes, int n_in,
                              void* d_out, int out_size, void* d_ws, size_t ws_size,
                              hipStream_t stream) {
  const float* x  = (const float*)d_in[0];  // [8,64,256,128]
  const float* W  = (const float*)d_in[1];  // [384,128]
  const float* b  = (const float*)d_in[2];  // [384]
  float*       o  = (float*)d_out;          // [8,64,256,128]

  const int nbl = in_sizes[0] / (G_ * C_);  // 8*64 = 512 (b,l) tiles
  dilated_msa_fused<<<dim3(nbl), dim3(256), 0, stream>>>(x, W, b, o);
}
